// Tube_fps_54425825574959
// MI455X (gfx1250) — matrix operand order repaired and checked by reading, not compiled, not run on hardware
//
#include <hip/hip_runtime.h>
#include <hip/hip_bf16.h>
#include <cmath>

typedef __attribute__((ext_vector_type(16))) _Float16 v16h;
typedef __attribute__((ext_vector_type(8)))  _Float16 v8h;
typedef __attribute__((ext_vector_type(8)))  float    v8f;

#define HDIM 768
#define SEQT 834
#define NBATCH 8
#define NROWS (NBATCH*SEQT)      // 6672
#define FFDIM 3072
#define TSEQ 1034
#define BN 64                    // block tile along N (4 x 16 per wave)

// ---------------------------------------------------------------------------
// fp16 WMMA GEMM:  C(M,N) = A(M,K) * W^T,  W stored row-major (N,K), fp16.
// Block = 256 threads = 8 waves stacked along M (block tile 128 x 64).
// Each wave computes a 16x64 strip: 4 accumulators, 4 v_wmma per k-step,
// A fragment reused across all 4. The 64x32 B tile is staged through LDS
// (double buffered, one barrier per k-step); rows >= N are zero-filled so
// WMMA always runs with EXEC all-ones and edge stores are simply guarded.
// A row for logical row m is A + (m+shift)*lda, valid iff (m%Trow)+shift in
// [0,Trow)  (per-batch shifted rows for the 9-tap temporal conv; zeros else).
// Epilogue row remap: crow = (m/Tin)*ldcb + cbase + (m%Tin)   (conv tubes
// scatter into the 834-token buffer; plain GEMMs use Tin=M, ldcb=0, cbase=0).
// flags: 1 = add bias[n], 2 = add R[crow*ldc+n] (residual / accumulate).
// ---------------------------------------------------------------------------
__global__ __launch_bounds__(256)
void gemm_f16(const _Float16* __restrict__ A, const _Float16* __restrict__ B,
              const float* __restrict__ bias, const float* __restrict__ R,
              float* __restrict__ C,
              int M, int N, int K, int lda, int ldb,
              int ldc, int Tin, int ldcb, int cbase,
              int shift, int Trow, int flags)
{
    __shared__ _Float16 lb[2][BN * 32];

    const int tid  = threadIdx.x;
    const int wave = tid >> 5;
    const int lane = tid & 31;
    const int r    = lane & 15;
    const int half = lane >> 4;

    const int mtile = (blockIdx.y * 8 + wave) * 16;
    const int ntile = blockIdx.x * BN;

    // ---- A gather setup (this lane's row) ----
    const int m  = mtile + r;
    const int tA = (m % Trow) + shift;
    const bool avalid = (m < M) && ((unsigned)tA < (unsigned)Trow);
    const _Float16* __restrict__ Arow = A + (long)(m + shift) * lda;

    // ---- B staging coords: thread -> one 16B chunk of the 64x32 tile ----
    const int sn = tid >> 2;          // 0..63 : row of tile (column of C)
    const int sk = (tid & 3) * 8;     // 0,8,16,24 : k offset
    const bool svalid = (ntile + sn) < N;
    const _Float16* __restrict__ Bst = B + (long)(ntile + sn) * ldb + sk;

    // prologue: stage k0 = 0 into buffer 0
    {
        v8h bv = {};
        if (svalid) bv = *(const v8h*)Bst;
        *(v8h*)(&lb[0][sn * 32 + sk]) = bv;
    }

    v8f acc[4] = {};
    const int nsteps = K / 32;
    for (int s = 0; s < nsteps; ++s) {
        __syncthreads();                       // tile s ready; tile s-1 consumed
        const int k0 = s * 32;
        if (s + 1 < nsteps) {                  // stage next tile into other buffer
            v8h bv = {};
            if (svalid) bv = *(const v8h*)(Bst + k0 + 32);
            *(v8h*)(&lb[(s + 1) & 1][sn * 32 + sk]) = bv;
        }
        // A fragment (two contiguous b128 loads; zeros for invalid rows)
        v16h af;
        if (avalid) {
            __builtin_prefetch(Arow + k0 + 64, 0, 3);
            v8h a0 = *(const v8h*)(Arow + k0 + half * 8);
            v8h a1 = *(const v8h*)(Arow + k0 + 16 + half * 8);
#pragma unroll
            for (int i = 0; i < 8; ++i) { af[i] = a0[i]; af[8 + i] = a1[i]; }
        } else {
#pragma unroll
            for (int i = 0; i < 16; ++i) af[i] = (_Float16)0.0f;
        }
        // 4 WMMAs against the staged B tile
        const _Float16* lbp = &lb[s & 1][0];
#pragma unroll
        for (int j = 0; j < 4; ++j) {
            const int nl = j * 16 + r;
            v8h b0 = *(const v8h*)(lbp + nl * 32 + half * 8);
            v8h b1 = *(const v8h*)(lbp + nl * 32 + 16 + half * 8);
            v16h bf;
#pragma unroll
            for (int i = 0; i < 8; ++i) { bf[i] = b0[i]; bf[8 + i] = b1[i]; }
            acc[j] = __builtin_amdgcn_wmma_f32_16x16x32_f16(
                         false, af, false, bf, (short)0, acc[j], false, false);
        }
    }

    // ---- epilogue: lane's column per subtile is ntile + j*16 + r ----
#pragma unroll
    for (int j = 0; j < 4; ++j) {
        const int n = ntile + j * 16 + r;
        if (n < N) {
            const float bv = (flags & 1) ? bias[n] : 0.0f;
#pragma unroll
            for (int v = 0; v < 8; ++v) {
                const int mr = mtile + v + half * 8;
                if (mr < M) {
                    const int crow = (mr / Tin) * ldcb + cbase + (mr % Tin);
                    const long idx = (long)crow * ldc + n;
                    float val = acc[j][v] + bv;
                    if (flags & 2) val += R[idx];
                    C[idx] = val;
                }
            }
        }
    }
}

// --------------------------- elementwise helpers ---------------------------
__global__ void cvt_f32_f16(const float* __restrict__ s, _Float16* __restrict__ d, long n)
{
    long i = (long)blockIdx.x * blockDim.x + threadIdx.x;
    if (i < n) d[i] = (_Float16)s[i];
}

__global__ void gelu_cvt(const float* __restrict__ s, _Float16* __restrict__ d, long n)
{
    long i = (long)blockIdx.x * blockDim.x + threadIdx.x;
    if (i < n) {
        float x = s[i];
        d[i] = (_Float16)(0.5f * x * (1.0f + erff(x * 0.70710678f)));
    }
}

// (O,C,9) f32  ->  tap-major (9,O,C) fp16, so every tap GEMM is contiguous
__global__ void repack_tap(const float* __restrict__ s, _Float16* __restrict__ d,
                           int O, int C)
{
    long i = (long)blockIdx.x * blockDim.x + threadIdx.x;
    long tot = (long)O * C * 9;
    if (i >= tot) return;
    int k  = (int)(i % 9);
    long oc = i / 9;                               // o*C + c
    d[(long)k * O * C + oc] = (_Float16)s[i];
}

// f32 (b,Tin,cols) -> f16 (b,outld,cols) at row offset `base`
__global__ void remap_cvt(const float* __restrict__ s, _Float16* __restrict__ d,
                          int nb, int Tin, int cols, int outld, int base)
{
    long i = (long)blockIdx.x * blockDim.x + threadIdx.x;
    long tot = (long)nb * Tin * cols;
    if (i >= tot) return;
    int c = (int)(i % cols);
    int t = (int)((i / cols) % Tin);
    int b = (int)(i / ((long)cols * Tin));
    d[((long)b * outld + base + t) * cols + c] = (_Float16)s[i];
}

// trilinear (half-pixel, PyTorch-style) resize of conv_w (768*3, 8,8,8) -> (OD,OH,OW)
__device__ inline void src_coord(int o, int O, int I, int& i0, int& i1, float& t)
{
    float f = ((float)o + 0.5f) * ((float)I / (float)O) - 0.5f;
    if (f < 0.0f) f = 0.0f;
    i0 = (int)floorf(f);
    if (i0 > I - 1) i0 = I - 1;
    t = f - (float)i0;
    i1 = i0 + 1; if (i1 > I - 1) i1 = I - 1;
}

__global__ void resize_w(const float* __restrict__ w, _Float16* __restrict__ out,
                         int OD, int OH, int OW)
{
    long i = (long)blockIdx.x * blockDim.x + threadIdx.x;
    long tot = (long)HDIM * 3 * OD * OH * OW;
    if (i >= tot) return;
    int ow = (int)(i % OW);
    int oh = (int)((i / OW) % OH);
    int od = (int)((i / ((long)OW * OH)) % OD);
    int oc = (int)(i / ((long)OW * OH * OD));
    int d0, d1, h0, h1, w0, w1; float td, th, tw;
    src_coord(od, OD, 8, d0, d1, td);
    src_coord(oh, OH, 8, h0, h1, th);
    src_coord(ow, OW, 8, w0, w1, tw);
    const float* p = w + (long)oc * 512;
    auto at = [&](int d, int h, int ww) { return p[(d * 8 + h) * 8 + ww]; };
    float c00 = at(d0, h0, w0) * (1 - tw) + at(d0, h0, w1) * tw;
    float c01 = at(d0, h1, w0) * (1 - tw) + at(d0, h1, w1) * tw;
    float c10 = at(d1, h0, w0) * (1 - tw) + at(d1, h0, w1) * tw;
    float c11 = at(d1, h1, w0) * (1 - tw) + at(d1, h1, w1) * tw;
    float c0 = c00 * (1 - th) + c01 * th;
    float c1 = c10 * (1 - th) + c11 * th;
    out[i] = (_Float16)(c0 * (1 - td) + c1 * td);
}

// im2col for one conv tube: x (8,3,64,224,224) -> col (8*ntok, K) fp16
__global__ void im2col3d(const float* __restrict__ x, _Float16* __restrict__ col,
                         int ntok, int K, int KD, int KH, int KW,
                         int SD, int SH, int SW, int OD, int OH, int OW)
{
    long i = (long)blockIdx.x * blockDim.x + threadIdx.x;
    long tot = (long)NBATCH * ntok * K;
    if (i >= tot) return;
    int kk = (int)(i % K);
    long m = i / K;
    int b = (int)(m / ntok), tk = (int)(m % ntok);
    int od = tk / (OH * OW); int rr = tk % (OH * OW);
    int oh = rr / OW, ow = rr % OW;
    int ksp = KD * KH * KW;
    int c = kk / ksp; int r2 = kk % ksp;
    int kd = r2 / (KH * KW); int r3 = r2 % (KH * KW);
    int kh = r3 / KW, kw = r3 % KW;
    int id = od * SD + kd, ih = oh * SH + kh, iw = ow * SW + kw;
    col[i] = (_Float16)x[(((long)(b * 3 + c) * 64 + id) * 224 + ih) * 224 + iw];
}

// cls token + positional embedding (conv GEMMs already wrote rows 1..833)
__global__ void pos_cls(float* __restrict__ h, const float* __restrict__ pos,
                        const float* __restrict__ cls)
{
    long i = (long)blockIdx.x * blockDim.x + threadIdx.x;
    long tot = (long)NBATCH * SEQT * HDIM;
    if (i >= tot) return;
    int c = (int)(i % HDIM);
    int t = (int)((i / HDIM) % SEQT);
    if (t == 0) h[i] = cls[c] + pos[c];
    else        h[i] += pos[(long)t * HDIM + c];
}

// GroupNorm(32 groups, eps 1e-6) over (channels-in-group x 834 tokens), fp16 out
__global__ __launch_bounds__(256)
void groupnorm(const float* __restrict__ h, _Float16* __restrict__ out,
               const float* __restrict__ g, const float* __restrict__ be)
{
    __shared__ float s1[256], s2[256];
    const int b = blockIdx.x >> 5, grp = blockIdx.x & 31;
    const int tid = threadIdx.x;
    const int CNT = 24 * SEQT;
    float a = 0.0f, q = 0.0f;
    for (int i = tid; i < CNT; i += 256) {
        int c = grp * 24 + (i % 24); int t = i / 24;
        float v = h[((long)b * SEQT + t) * HDIM + c];
        a += v; q += v * v;
    }
    s1[tid] = a; s2[tid] = q; __syncthreads();
    for (int s = 128; s > 0; s >>= 1) {
        if (tid < s) { s1[tid] += s1[tid + s]; s2[tid] += s2[tid + s]; }
        __syncthreads();
    }
    const float mean = s1[0] / (float)CNT;
    const float var  = s2[0] / (float)CNT - mean * mean;
    const float rstd = rsqrtf(var + 1e-6f);
    for (int i = tid; i < CNT; i += 256) {
        int c = grp * 24 + (i % 24); int t = i / 24;
        long idx = ((long)b * SEQT + t) * HDIM + c;
        out[idx] = (_Float16)((h[idx] - mean) * rstd * g[c] + be[c]);
    }
}

// LayerNorm(768, eps 1e-5), fp16 out; one block per row
__global__ __launch_bounds__(256)
void layernorm(const float* __restrict__ x, _Float16* __restrict__ out,
               const float* __restrict__ g, const float* __restrict__ be)
{
    __shared__ float s1[256], s2[256];
    const long row = blockIdx.x;
    const float* xr = x + row * HDIM;
    const int tid = threadIdx.x;
    float a = 0.0f, q = 0.0f;
    for (int c = tid; c < HDIM; c += 256) { float v = xr[c]; a += v; q += v * v; }
    s1[tid] = a; s2[tid] = q; __syncthreads();
    for (int s = 128; s > 0; s >>= 1) {
        if (tid < s) { s1[tid] += s1[tid + s]; s2[tid] += s2[tid + s]; }
        __syncthreads();
    }
    const float mean = s1[0] / (float)HDIM;
    const float rstd = rsqrtf(s2[0] / (float)HDIM - mean * mean + 1e-5f);
    for (int c = tid; c < HDIM; c += 256)
        out[row * HDIM + c] = (_Float16)((xr[c] - mean) * rstd * g[c] + be[c]);
}

// One query row of one head: full softmax over 834 keys, DH=48
__global__ __launch_bounds__(64)
void flash_attn(const float* __restrict__ q, const float* __restrict__ k,
                const float* __restrict__ v, float* __restrict__ o)
{
    __shared__ float sc[SEQT];
    __shared__ float qv[48];
    __shared__ float red[64];
    const int i = blockIdx.x, hh = blockIdx.y, b = blockIdx.z;
    const int tid = threadIdx.x;
    const long base = (long)b * SEQT * HDIM + hh * 48;
    if (tid < 48) qv[tid] = q[base + (long)i * HDIM + tid];
    __syncthreads();
    for (int j = tid; j < SEQT; j += 64) {
        const float* kr = k + base + (long)j * HDIM;
        float s = 0.0f;
#pragma unroll 4
        for (int d = 0; d < 48; ++d) s += qv[d] * kr[d];
        sc[j] = s * 0.14433757f; // 48^-0.5
    }
    __syncthreads();
    float mx = -1e30f;
    for (int j = tid; j < SEQT; j += 64) mx = fmaxf(mx, sc[j]);
    red[tid] = mx; __syncthreads();
    for (int s = 32; s > 0; s >>= 1) { if (tid < s) red[tid] = fmaxf(red[tid], red[tid + s]); __syncthreads(); }
    mx = red[0]; __syncthreads();
    float sm = 0.0f;
    for (int j = tid; j < SEQT; j += 64) { float p = __expf(sc[j] - mx); sc[j] = p; sm += p; }
    red[tid] = sm; __syncthreads();
    for (int s = 32; s > 0; s >>= 1) { if (tid < s) red[tid] += red[tid + s]; __syncthreads(); }
    const float inv = 1.0f / red[0];
    __syncthreads();
    if (tid < 48) {
        const float* vc = v + base + tid;
        float acc = 0.0f;
        for (int j = 0; j < SEQT; ++j) acc += sc[j] * vc[(long)j * HDIM];
        o[base + (long)i * HDIM + tid] = acc * inv;
    }
}

// GRU scan (torch cell) + linear head + masked mean. One block: 8 batches x 32 units.
__global__ __launch_bounds__(256)
void gru_head(const float* __restrict__ gx, const float* __restrict__ whh,
              const float* __restrict__ bhh, const float* __restrict__ qw,
              const float* __restrict__ qb, const int* __restrict__ flen,
              float* __restrict__ out)
{
    __shared__ float hs[256], ps[256];
    const int tid = threadIdx.x;
    const int b = tid >> 5, u = tid & 31;
    hs[tid] = 0.0f; __syncthreads();
    float acc = 0.0f;
    const int L = flen[b] + SEQT; // fea_len + 834
    for (int t = 0; t < TSEQ; ++t) {
        const float* hb = hs + b * 32;
        float gr = bhh[u], gz = bhh[32 + u], gn = bhh[64 + u];
#pragma unroll 4
        for (int j = 0; j < 32; ++j) {
            float hv = hb[j];
            gr += whh[u * 32 + j] * hv;
            gz += whh[(32 + u) * 32 + j] * hv;
            gn += whh[(64 + u) * 32 + j] * hv;
        }
        const float* gt = gx + ((long)b * TSEQ + t) * 96;
        float rr = 1.0f / (1.0f + __expf(-(gt[u] + gr)));
        float zg = 1.0f / (1.0f + __expf(-(gt[32 + u] + gz)));
        float nn = tanhf(gt[64 + u] + rr * gn);
        float hn = (1.0f - zg) * nn + zg * hs[tid];
        __syncthreads();
        hs[tid] = hn; ps[tid] = hn * qw[u];
        __syncthreads();
        if (u == 0 && t < L) {
            float s = qb[0];
            for (int j = 0; j < 32; ++j) s += ps[b * 32 + j];
            acc += s;
        }
    }
    if (u == 0) out[b] = acc / (float)L;
}

// ---------------------------------------------------------------------------
extern "C" void kernel_launch(void* const* d_in, const int* in_sizes, int n_in,
                              void* d_out, int out_size, void* d_ws, size_t ws_size,
                              hipStream_t stream)
{
    const float* x        = (const float*)d_in[0];
    const float* spa_feas = (const float*)d_in[1];
    const int*   fea_len  = (const int*)  d_in[2];
    const float* pos_emb  = (const float*)d_in[3];
    const float* cls_tok  = (const float*)d_in[4];
    const float* conv_w   = (const float*)d_in[5];
    const float* conv_b   = (const float*)d_in[6];
    const float* gn_g     = (const float*)d_in[7];
    const float* gn_b     = (const float*)d_in[8];
    const float* pin_w    = (const float*)d_in[9];
    const float* pin_b    = (const float*)d_in[10];
    const float* ln_g[3]  = {(const float*)d_in[11], (const float*)d_in[13], (const float*)d_in[15]};
    const float* ln_b[3]  = {(const float*)d_in[12], (const float*)d_in[14], (const float*)d_in[16]};
    const float* awq[2]   = {(const float*)d_in[17], (const float*)d_in[22]};
    const float* awk[2]   = {(const float*)d_in[18], (const float*)d_in[23]};
    const float* awv[2]   = {(const float*)d_in[19], (const float*)d_in[24]};
    const float* awo[2]   = {(const float*)d_in[20], (const float*)d_in[25]};
    const float* abo[2]   = {(const float*)d_in[21], (const float*)d_in[26]};
    const float* ff_w1    = (const float*)d_in[27];
    const float* ff_b1    = (const float*)d_in[28];
    const float* ff_w2    = (const float*)d_in[29];
    const float* ff_b2    = (const float*)d_in[30];
    const float* pout_w   = (const float*)d_in[31];
    const float* pout_b   = (const float*)d_in[32];
    const float* spa_w    = (const float*)d_in[33];
    const float* spa_b    = (const float*)d_in[34];
    const float* gru_wih  = (const float*)d_in[35];
    const float* gru_whh  = (const float*)d_in[36];
    const float* gru_bih  = (const float*)d_in[37];
    const float* gru_bhh  = (const float*)d_in[38];
    const float* qh_w     = (const float*)d_in[39];
    const float* qh_b     = (const float*)d_in[40];
    float* outp = (float*)d_out;

    // bump allocator over workspace (256B aligned)
    char* wsb = (char*)d_ws; size_t off = 0;
    auto alloc = [&](size_t bytes) -> void* {
        void* p = wsb + off; off += (bytes + 255) & ~(size_t)255; return p;
    };
    const long ROWF = (long)NROWS * HDIM;      // 6672*768
    float*     h     = (float*)alloc(ROWF * 4);
    float*     z     = (float*)alloc(ROWF * 4);
    float*     qb_   = (float*)alloc(4 * ROWF * 4);  // q,k,v,o ; alias: FFN mid (6672x3072)
    float*     kb_   = qb_ + ROWF;
    float*     vb_   = kb_ + ROWF;
    float*     ob_   = vb_ + ROWF;
    float*     mid   = qb_;
    _Float16*  act16 = (_Float16*)alloc(ROWF * 2);
    _Float16*  mid16 = (_Float16*)alloc((long)NROWS * FFDIM * 2);
    _Float16*  col16 = (_Float16*)alloc(2709504L * 2);
    _Float16*  w0_16 = (_Float16*)alloc(768L * 1536 * 2);
    _Float16*  w1t16 = (_Float16*)alloc(768L * 768 * 2);
    _Float16*  w2t16 = (_Float16*)alloc(768L * 1728 * 2);
    _Float16*  pin16 = (_Float16*)alloc(768L * 768 * 2);
    _Float16*  wq16[2], *wk16[2], *wv16[2], *wo16[2];
    for (int a = 0; a < 2; ++a) {
        wq16[a] = (_Float16*)alloc(768L * 768 * 2);
        wk16[a] = (_Float16*)alloc(768L * 768 * 2);
        wv16[a] = (_Float16*)alloc(768L * 768 * 2);
        wo16[a] = (_Float16*)alloc(768L * 768 * 2);
    }
    _Float16*  pout16 = (_Float16*)alloc(768L * 768 * 2);
    _Float16*  fw1r16 = (_Float16*)alloc((long)FFDIM * 768 * 9 * 2);  // (9,3072,768)
    _Float16*  fw2r16 = (_Float16*)alloc((long)HDIM * FFDIM * 9 * 2); // (9,768,3072)
    _Float16*  spaw16 = (_Float16*)alloc(768L * 2048 * 2);
    _Float16*  wih16  = (_Float16*)alloc(96L * 768 * 2);
    _Float16*  spain16= (_Float16*)alloc((long)NBATCH * 200 * 2048 * 2);
    float*     spaF   = (float*)alloc((long)NBATCH * 200 * 768 * 4);
    _Float16*  seq16  = (_Float16*)alloc((long)NBATCH * TSEQ * 768 * 2);
    float*     gx     = (float*)alloc((long)NBATCH * TSEQ * 96 * 4);
    (void)ws_size; (void)in_sizes; (void)n_in; (void)out_size;

    auto cvt = [&](const float* s, _Float16* d, long n) {
        cvt_f32_f16<<<dim3((unsigned)((n + 255) / 256)), 256, 0, stream>>>(s, d, n);
    };
    auto gemm = [&](const _Float16* A, const _Float16* B, const float* bias,
                    const float* R, float* C, int M, int N, int K, int lda,
                    int ldb, int ldc, int Tin, int ldcb, int cbase,
                    int shift, int Trow, int flags) {
        dim3 g((unsigned)((N + BN - 1) / BN), (unsigned)((M + 127) / 128));
        gemm_f16<<<g, 256, 0, stream>>>(A, B, bias, R, C, M, N, K, lda, ldb,
                                        ldc, Tin, ldcb, cbase, shift, Trow, flags);
    };

    // ---- weight conversions / repacks (every call; deterministic) ----
    cvt(conv_w, w0_16, 768L * 3 * 512);
    resize_w<<<dim3((unsigned)((768L*3*256 + 255)/256)), 256, 0, stream>>>(conv_w, w1t16, 16, 4, 4);
    resize_w<<<dim3((unsigned)((768L*3*576 + 255)/256)), 256, 0, stream>>>(conv_w, w2t16, 4, 12, 12);
    cvt(pin_w, pin16, 768L * 768);
    for (int a = 0; a < 2; ++a) {
        cvt(awq[a], wq16[a], 768L * 768); cvt(awk[a], wk16[a], 768L * 768);
        cvt(awv[a], wv16[a], 768L * 768); cvt(awo[a], wo16[a], 768L * 768);
    }
    cvt(pout_w, pout16, 768L * 768);
    repack_tap<<<dim3((unsigned)(((long)FFDIM*768*9 + 255)/256)), 256, 0, stream>>>(
        ff_w1, fw1r16, FFDIM, HDIM);
    repack_tap<<<dim3((unsigned)(((long)HDIM*FFDIM*9 + 255)/256)), 256, 0, stream>>>(
        ff_w2, fw2r16, HDIM, FFDIM);
    cvt(spa_w, spaw16, 768L * 2048);
    cvt(gru_wih, wih16, 96L * 768);
    cvt(spa_feas, spain16, (long)NBATCH * 200 * 2048);

    // ---- tokenizer: 3 conv tubes as im2col + WMMA GEMM, scatter into h ----
    struct Tube { int KD,KH,KW,SD,SH,SW,OD,ntok,K,base; _Float16* w; };
    Tube tubes[3] = {
        { 8, 8, 8,16,32,32, 4, 196, 1536,   0, w0_16 },
        {16, 4, 4, 6,32,32, 9, 441,  768, 196, w1t16 },
        { 4,12,12,16,32,32, 4, 196, 1728, 637, w2t16 },
    };
    for (int i = 0; i < 3; ++i) {
        Tube& T = tubes[i];
        long tot = (long)NBATCH * T.ntok * T.K;
        im2col3d<<<dim3((unsigned)((tot + 255) / 256)), 256, 0, stream>>>(
            x, col16, T.ntok, T.K, T.KD, T.KH, T.KW, T.SD, T.SH, T.SW, T.OD, 7, 7);
        int M = NBATCH * T.ntok;
        gemm(col16, T.w, conv_b + i * HDIM, nullptr, h, M, HDIM, T.K, T.K, T.K,
             HDIM, T.ntok, SEQT, 1 + T.base, 0, M, 1);
    }
    pos_cls<<<dim3((unsigned)((ROWF + 255) / 256)), 256, 0, stream>>>(h, pos_emb, cls_tok);

    // ---- temporal encoder ----
    groupnorm<<<dim3(NBATCH * 32), 256, 0, stream>>>(h, act16, gn_g, gn_b);
    gemm(act16, pin16, pin_b, nullptr, z, NROWS, HDIM, HDIM, HDIM, HDIM,
         HDIM, NROWS, 0, 0, 0, NROWS, 1);

    // two attention blocks
    for (int a = 0; a < 2; ++a) {
        layernorm<<<dim3(NROWS), 256, 0, stream>>>(z, act16, ln_g[a], ln_b[a]);
        gemm(act16, wq16[a], nullptr, nullptr, qb_, NROWS, HDIM, HDIM, HDIM,
             HDIM, HDIM, NROWS, 0, 0, 0, NROWS, 0);
        gemm(act16, wk16[a], nullptr, nullptr, kb_, NROWS, HDIM, HDIM, HDIM,
             HDIM, HDIM, NROWS, 0, 0, 0, NROWS, 0);
        gemm(act16, wv16[a], nullptr, nullptr, vb_, NROWS, HDIM, HDIM, HDIM,
             HDIM, HDIM, NROWS, 0, 0, 0, NROWS, 0);
        flash_attn<<<dim3(SEQT, 16, NBATCH), 64, 0, stream>>>(qb_, kb_, vb_, ob_);
        cvt(ob_, act16, ROWF);
        gemm(act16, wo16[a], abo[a], z, z, NROWS, HDIM, HDIM, HDIM, HDIM,
             HDIM, NROWS, 0, 0, 0, NROWS, 3);
    }

    // FFN: ln3 -> conv1d(k=9,pad=4) -> GELU -> conv1d(k=9,pad=4) -> +z
    layernorm<<<dim3(NROWS), 256, 0, stream>>>(z, act16, ln_g[2], ln_b[2]);
    for (int k = 0; k < 9; ++k)
        gemm(act16, fw1r16 + (long)k * FFDIM * HDIM, ff_b1,
             (k == 0 ? nullptr : mid), mid,
             NROWS, FFDIM, HDIM, HDIM, HDIM, FFDIM,
             NROWS, 0, 0, k - 4, SEQT, (k == 0 ? 1 : 2));
    gelu_cvt<<<dim3((unsigned)(((long)NROWS*FFDIM + 255)/256)), 256, 0, stream>>>(
        mid, mid16, (long)NROWS * FFDIM);
    for (int k = 0; k < 9; ++k)
        gemm(mid16, fw2r16 + (long)k * HDIM * FFDIM, ff_b2, z, z,
             NROWS, HDIM, FFDIM, FFDIM, FFDIM, HDIM,
             NROWS, 0, 0, k - 4, SEQT, (k == 0 ? 3 : 2));

    // pout + residual with x_in (= h)
    cvt(z, act16, ROWF);
    gemm(act16, pout16, pout_b, h, h, NROWS, HDIM, HDIM, HDIM, HDIM,
         HDIM, NROWS, 0, 0, 0, NROWS, 3);

    // ---- seq = concat(h, spa) as fp16 ----
    remap_cvt<<<dim3((unsigned)((ROWF + 255)/256)), 256, 0, stream>>>(
        h, seq16, NBATCH, SEQT, HDIM, TSEQ, 0);
    gemm(spain16, spaw16, spa_b, nullptr, spaF, NBATCH * 200, HDIM, 2048, 2048,
         2048, HDIM, NBATCH * 200, 0, 0, 0, NBATCH * 200, 1);
    remap_cvt<<<dim3((unsigned)(((long)NBATCH*200*HDIM + 255)/256)), 256, 0, stream>>>(
        spaF, seq16, NBATCH, 200, HDIM, TSEQ, SEQT);

    // ---- GRU input GEMM + scan + head ----
    gemm(seq16, wih16, gru_bih, nullptr, gx, NBATCH * TSEQ, 96, HDIM, HDIM,
         HDIM, 96, NBATCH * TSEQ, 0, 0, 0, NBATCH * TSEQ, 1);
    gru_head<<<dim3(1), 256, 0, stream>>>(gx, gru_whh, gru_bhh, qh_w, qh_b,
                                          fea_len, outp);
}
